// EquivariantGraphNeuralOperatorBlock_1090921693625
// MI455X (gfx1250) — matrix operand order repaired and checked by reading, not compiled, not run on hardware
//
#include <hip/hip_runtime.h>

// ---------------------------------------------------------------------------
// EquivariantGraphNeuralOperatorBlock for MI455X (gfx1250, wave32, WMMA)
// bf16 WMMA (f32 accumulate) for all edge/node MLP GEMMs.
// Weights kept TRANSPOSED ([out][K]) so B-fragments are contiguous 16B loads.
// ---------------------------------------------------------------------------

#define T_      8
#define N_      10000
#define F_      64
#define E_      640000
#define NODES_  (T_ * N_)        // 80000
#define H_      64
#define ED_     16
#define KIN_    145              // 2F + 1 + ED
#define KINP_   160              // padded to multiple of 32

typedef __bf16 v16bf __attribute__((ext_vector_type(16)));
typedef __bf16 v8bf  __attribute__((ext_vector_type(8)));
typedef float  v8f   __attribute__((ext_vector_type(8)));

__device__ __forceinline__ float silu_f(float x) {
    return x / (1.0f + __expf(-x));
}

// --- WMMA fragment loaders (LDS bf16) --------------------------------------
// A: 16x32 tile, base points at A[row0][k0], lda = row stride (elements).
// ISA layout: lanes 0-15: M=lane, K = {0..7, 16..23}; lanes 16-31: M=lane-16,
// K = {8..15, 24..31}.  -> two contiguous 16B chunks per lane.
__device__ __forceinline__ v16bf load_frag_A(const __bf16* base, int lda, int lane) {
    int m  = lane & 15;
    int h8 = ((lane >> 4) & 1) * 8;
    const __bf16* row = base + m * lda + h8;
    v8bf lo = *(const v8bf*)(row);        // K = h8 .. h8+7
    v8bf hi = *(const v8bf*)(row + 16);   // K = 16+h8 .. 16+h8+7
    return __builtin_shufflevector(lo, hi, 0,1,2,3,4,5,6,7,8,9,10,11,12,13,14,15);
}

// B from TRANSPOSED weights BT[64][ldk] (= W^T row-major).
// ISA B layout: lanes 0-15 -> N=lane, K=kk+{0..7,16..23}; lanes 16-31 -> N=lane-16,
// K=kk+{8..15,24..31}.  Per-lane data = two 16B chunks of BT row.
__device__ __forceinline__ v16bf load_frag_BT(const __bf16* BT, int ldk,
                                              int nbase, int kk, int lane) {
    int n    = nbase + (lane & 15);
    int koff = kk + ((lane >> 4) << 3);
    const __bf16* row = BT + n * ldk + koff;
    v8bf lo = *(const v8bf*)(row);
    v8bf hi = *(const v8bf*)(row + 16);
    return __builtin_shufflevector(lo, hi, 0,1,2,3,4,5,6,7,8,9,10,11,12,13,14,15);
}

// --- 128xK * Kx64 GEMM done by one 256-thread (8-wave) block ---------------
// A in LDS (bf16, row-major, lda), BT in LDS [64][K] bf16 (transposed weight),
// bias[64] in LDS f32.  Each wave computes a 2x2 block of 16x16 tiles.
// Optional outputs: Obuf (LDS bf16 [128][64]), aggM (global atomic scatter by
// dstIdx[m]), gOut (global f32, rows gBase+m).
__device__ __forceinline__ void gemm128(
    const __bf16* __restrict__ A, int lda,
    const __bf16* __restrict__ BT,
    const float*  __restrict__ bias,
    int K, bool act,
    __bf16* Obuf,
    float* aggM, const int* dstIdx,
    float* gOut, int gBase,
    int tid)
{
    int wv   = tid >> 5;
    int lane = tid & 31;
    int n0   = lane & 15;
    int hf   = (lane >> 4) & 1;
    int mtb  = (wv >> 1) * 2;   // 0,2,4,6
    int ntb  = (wv & 1) * 2;    // 0,2

    v8f acc00 = {0.f,0.f,0.f,0.f,0.f,0.f,0.f,0.f};
    v8f acc01 = acc00, acc10 = acc00, acc11 = acc00;

    for (int kk = 0; kk < K; kk += 32) {
        v16bf a0 = load_frag_A(A + (mtb * 16) * lda + kk, lda, lane);
        v16bf a1 = load_frag_A(A + ((mtb + 1) * 16) * lda + kk, lda, lane);
        v16bf b0 = load_frag_BT(BT, K, ntb * 16, kk, lane);
        v16bf b1 = load_frag_BT(BT, K, (ntb + 1) * 16, kk, lane);
        acc00 = __builtin_amdgcn_wmma_f32_16x16x32_bf16(false, a0, false, b0, (short)0, acc00, false, false);
        acc01 = __builtin_amdgcn_wmma_f32_16x16x32_bf16(false, a0, false, b1, (short)0, acc01, false, false);
        acc10 = __builtin_amdgcn_wmma_f32_16x16x32_bf16(false, a1, false, b0, (short)0, acc10, false, false);
        acc11 = __builtin_amdgcn_wmma_f32_16x16x32_bf16(false, a1, false, b1, (short)0, acc11, false, false);
    }

#pragma unroll
    for (int ti = 0; ti < 4; ++ti) {
        int mt = mtb + (ti >> 1);
        int nt = ntb + (ti & 1);
        v8f acc = (ti == 0) ? acc00 : (ti == 1) ? acc01 : (ti == 2) ? acc10 : acc11;
        int n = nt * 16 + n0;
        float bv = bias[n];
#pragma unroll
        for (int r = 0; r < 8; ++r) {
            int m = mt * 16 + r + 8 * hf;
            float v = acc[r] + bv;
            if (act) v = silu_f(v);
            if (Obuf) Obuf[m * 64 + n] = (__bf16)v;
            if (aggM) atomicAdd(&aggM[(size_t)dstIdx[m] * 64 + n], v);
            if (gOut) gOut[(size_t)(gBase + m) * 64 + n] = v;
        }
    }
}

// ---------------------------------------------------------------------------
// Utility kernels
// ---------------------------------------------------------------------------
__global__ void zero_kernel(float* p, int n) {
    int i = blockIdx.x * 256 + threadIdx.x;
    if (i < n) p[i] = 0.f;
}

__global__ void cvt_kernel(const float* __restrict__ s, __bf16* __restrict__ d, int n) {
    int i = blockIdx.x * 256 + threadIdx.x;
    if (i < n) d[i] = (__bf16)s[i];
}

// Transpose weight [K][64] f32 -> [64][Kpad] bf16 (zero pad rows K..Kpad-1).
__global__ void cvt_t_kernel(const float* __restrict__ s, __bf16* __restrict__ d,
                             int K, int Kpad) {
    int i = blockIdx.x * 256 + threadIdx.x;
    if (i < 64 * Kpad) {
        int o = i / Kpad, k = i - o * Kpad;
        d[i] = (k < K) ? (__bf16)s[(size_t)k * 64 + o] : (__bf16)0.f;
    }
}

// center[t][d] = mean over n of pos[t][n][d]; one block per (t,d)
__global__ void center_kernel(const float* __restrict__ pos, float* __restrict__ center) {
    __shared__ float red[256];
    int t = blockIdx.x / 3, d = blockIdx.x % 3;
    float s = 0.f;
    for (int n = threadIdx.x; n < N_; n += 256)
        s += pos[((size_t)t * N_ + n) * 3 + d];
    red[threadIdx.x] = s;
    __syncthreads();
    for (int w = 128; w > 0; w >>= 1) {
        if (threadIdx.x < w) red[threadIdx.x] += red[threadIdx.x + w];
        __syncthreads();
    }
    if (threadIdx.x == 0) center[t * 3 + d] = red[0] * (1.0f / (float)N_);
}

// ---------------------------------------------------------------------------
// Spectral conv on x (FNO over T=8, 5 modes), one block per node n.
// ---------------------------------------------------------------------------
__global__ __launch_bounds__(256) void spectral_x_kernel(
    const float* __restrict__ x,
    const float* __restrict__ wsr, const float* __restrict__ wsi,
    float* __restrict__ xs_f, __bf16* __restrict__ xs_bf)
{
    __shared__ float Fr[5][64], Fi[5][64], Gr[5][64], Gi[5][64];
    int n = blockIdx.x;
    int tid = threadIdx.x;
    const float PI4 = 0.78539816339744831f;   // 2*pi/8

    if (tid < 64) {
        float xv[8];
#pragma unroll
        for (int t = 0; t < 8; ++t)
            xv[t] = x[((size_t)t * N_ + n) * 64 + tid];
#pragma unroll
        for (int k = 0; k < 5; ++k) {
            float fr = 0.f, fi = 0.f;
#pragma unroll
            for (int t = 0; t < 8; ++t) {
                float ang = PI4 * (float)(k * t);
                fr += xv[t] * __cosf(ang);
                fi -= xv[t] * __sinf(ang);
            }
            Fr[k][tid] = fr; Fi[k][tid] = fi;
        }
    }
    __syncthreads();
    for (int idx = tid; idx < 5 * 64; idx += 256) {
        int k = idx >> 6, o = idx & 63;
        float gr = 0.f, gi = 0.f;
        for (int i = 0; i < 64; ++i) {
            float wr = wsr[((size_t)i * 64 + o) * 5 + k];
            float wi = wsi[((size_t)i * 64 + o) * 5 + k];
            float fr = Fr[k][i], fi = Fi[k][i];
            gr += fr * wr - fi * wi;
            gi += fr * wi + fi * wr;
        }
        Gr[k][o] = gr; Gi[k][o] = gi;
    }
    __syncthreads();
    for (int idx = tid; idx < 8 * 64; idx += 256) {
        int t = idx >> 6, o = idx & 63;
        float s = Gr[0][o];
#pragma unroll
        for (int k = 1; k < 4; ++k) {
            float ang = PI4 * (float)(k * t);
            s += 2.f * (Gr[k][o] * __cosf(ang) - Gi[k][o] * __sinf(ang));
        }
        s += ((t & 1) ? -1.f : 1.f) * Gr[4][o];
        s *= 0.125f;
        size_t node = (size_t)t * N_ + n;
        float y = x[node * 64 + o] + s;
        xs_f [node * 64 + o] = y;
        xs_bf[node * 64 + o] = (__bf16)y;
    }
}

// ---------------------------------------------------------------------------
// Spectral conv on vector = stack(pos - center, vel), 2x2 mode weights.
// ---------------------------------------------------------------------------
__global__ void spectral_vec_kernel(
    const float* __restrict__ pos, const float* __restrict__ vel,
    const float* __restrict__ center,
    const float* __restrict__ wvr, const float* __restrict__ wvi,
    float* __restrict__ pos2, float* __restrict__ vel2)
{
    int idx = blockIdx.x * 256 + threadIdx.x;
    if (idx >= N_ * 3) return;
    int n = idx / 3, d = idx % 3;
    const float PI4 = 0.78539816339744831f;

    float sig[2][8];
#pragma unroll
    for (int t = 0; t < 8; ++t) {
        size_t off = ((size_t)t * N_ + n) * 3 + d;
        sig[0][t] = pos[off] - center[t * 3 + d];
        sig[1][t] = vel[off];
    }
    float Fr[2][5], Fi[2][5];
#pragma unroll
    for (int c = 0; c < 2; ++c)
#pragma unroll
        for (int k = 0; k < 5; ++k) {
            float fr = 0.f, fi = 0.f;
#pragma unroll
            for (int t = 0; t < 8; ++t) {
                float ang = PI4 * (float)(k * t);
                fr += sig[c][t] * __cosf(ang);
                fi -= sig[c][t] * __sinf(ang);
            }
            Fr[c][k] = fr; Fi[c][k] = fi;
        }
    float Gr[2][5], Gi[2][5];
#pragma unroll
    for (int o = 0; o < 2; ++o)
#pragma unroll
        for (int k = 0; k < 5; ++k) {
            float gr = 0.f, gi = 0.f;
#pragma unroll
            for (int i = 0; i < 2; ++i) {
                float wr = wvr[((size_t)i * 2 + o) * 5 + k];
                float wi = wvi[((size_t)i * 2 + o) * 5 + k];
                gr += Fr[i][k] * wr - Fi[i][k] * wi;
                gi += Fr[i][k] * wi + Fi[i][k] * wr;
            }
            Gr[o][k] = gr; Gi[o][k] = gi;
        }
#pragma unroll
    for (int t = 0; t < 8; ++t) {
        float out[2];
#pragma unroll
        for (int o = 0; o < 2; ++o) {
            float s = Gr[o][0];
#pragma unroll
            for (int k = 1; k < 4; ++k) {
                float ang = PI4 * (float)(k * t);
                s += 2.f * (Gr[o][k] * __cosf(ang) - Gi[o][k] * __sinf(ang));
            }
            s += ((t & 1) ? -1.f : 1.f) * Gr[o][4];
            out[o] = s * 0.125f;
        }
        size_t off = ((size_t)t * N_ + n) * 3 + d;
        pos2[off] = pos[off] + out[0];
        vel2[off] = vel[off] + out[1];
    }
}

// ---------------------------------------------------------------------------
// Edge kernel: 128 edges per block.  message MLP (3 GEMMs) + upd_pos MLP
// (2 GEMMs + dot head), scatter-add agg_m / agg_pos.
// ---------------------------------------------------------------------------
__global__ __launch_bounds__(256) void edge_kernel(
    const int*   __restrict__ eidx,       // [2][E]
    const float* __restrict__ edge_attr,  // [E][16]
    const __bf16* __restrict__ xs_bf,     // [NODES][64]
    const float* __restrict__ pos2,       // [NODES][3]
    const __bf16* __restrict__ wm1, const __bf16* __restrict__ wm2,
    const __bf16* __restrict__ wm3,
    const __bf16* __restrict__ wp1, const __bf16* __restrict__ wp2,
    const __bf16* __restrict__ wp3g,
    const float* __restrict__ bm1, const float* __restrict__ bm2,
    const float* __restrict__ bm3,
    const float* __restrict__ bp1, const float* __restrict__ bp2,
    const float* __restrict__ bp3,
    float* __restrict__ agg_m,            // [NODES][64]
    float* __restrict__ agg_pos)          // [NODES][3]
{
    extern __shared__ char smem[];
    char* p = smem;
    float* sb1   = (float*)p; p += 256;
    float* sb2   = (float*)p; p += 256;
    float* sb3   = (float*)p; p += 256;
    float* sbp1  = (float*)p; p += 256;
    float* sbp2  = (float*)p; p += 256;
    float* sdiff = (float*)p; p += 128 * 3 * 4;
    float* sdist = (float*)p; p += 128 * 4;
    int*   ssrc  = (int*)p;   p += 128 * 4;
    int*   sdst  = (int*)p;   p += 128 * 4;
    __bf16* inA  = (__bf16*)p; p += 128 * KINP_ * 2;
    __bf16* sW1  = (__bf16*)p; p += KINP_ * 64 * 2;  // [64][160] transposed
    __bf16* sW2  = (__bf16*)p; p += 4096 * 2;        // [64][64] transposed
    __bf16* sW3  = (__bf16*)p; p += 4096 * 2;
    __bf16* sWp1 = (__bf16*)p; p += 4096 * 2;
    __bf16* sWp2 = (__bf16*)p; p += 4096 * 2;
    __bf16* swp3 = (__bf16*)p; p += 64 * 2;
    __bf16* hB   = (__bf16*)p; p += 128 * 64 * 2;
    __bf16* hC   = (__bf16*)p; p += 128 * 64 * 2;
    __bf16* hD   = (__bf16*)p; p += 128 * 64 * 2;

    int tid = threadIdx.x;
    int e0  = blockIdx.x * 128;

    // --- stage weights / biases into LDS (vectorized copies) ---
    {
        const v8bf* s8; v8bf* d8;
        s8 = (const v8bf*)wm1; d8 = (v8bf*)sW1;
        for (int i = tid; i < KINP_ * 64 / 8; i += 256) d8[i] = s8[i];
        for (int i = tid; i < 4096 / 8; i += 256) {
            ((v8bf*)sW2)[i]  = ((const v8bf*)wm2)[i];
            ((v8bf*)sW3)[i]  = ((const v8bf*)wm3)[i];
            ((v8bf*)sWp1)[i] = ((const v8bf*)wp1)[i];
            ((v8bf*)sWp2)[i] = ((const v8bf*)wp2)[i];
        }
    }
    if (tid < 64) {
        swp3[tid] = wp3g[tid];
        sb1[tid] = bm1[tid]; sb2[tid] = bm2[tid]; sb3[tid] = bm3[tid];
        sbp1[tid] = bp1[tid]; sbp2[tid] = bp2[tid];
    }

    // --- per-edge geometry ---
    if (tid < 128) {
        int e = e0 + tid;
        int s = eidx[e], d = eidx[E_ + e];
        ssrc[tid] = s; sdst[tid] = d;
        float dx = pos2[(size_t)d * 3 + 0] - pos2[(size_t)s * 3 + 0];
        float dy = pos2[(size_t)d * 3 + 1] - pos2[(size_t)s * 3 + 1];
        float dz = pos2[(size_t)d * 3 + 2] - pos2[(size_t)s * 3 + 2];
        sdiff[tid * 3 + 0] = dx; sdiff[tid * 3 + 1] = dy; sdiff[tid * 3 + 2] = dz;
        sdist[tid] = sqrtf(dx * dx + dy * dy + dz * dz);
    }
    __syncthreads();

    // --- build input tile [128][160] bf16: [x[dst] | x[src] | dist | eattr | 0]
    {
        int r  = tid >> 1;
        int hb = tid & 1;                  // 0 -> x[dst], 1 -> x[src]
        int e  = e0 + r;
        int node = hb ? ssrc[r] : sdst[r];
        const v8bf* srow = (const v8bf*)(xs_bf + (size_t)node * 64);
        v8bf* drow = (v8bf*)(inA + r * KINP_ + hb * 64);
#pragma unroll
        for (int j = 0; j < 8; ++j) drow[j] = srow[j];
        if (hb == 0) {
            inA[r * KINP_ + 128] = (__bf16)sdist[r];
#pragma unroll
            for (int c = 0; c < 16; ++c)
                inA[r * KINP_ + 129 + c] = (__bf16)edge_attr[(size_t)e * 16 + c];
#pragma unroll
            for (int c = 145; c < 160; ++c)
                inA[r * KINP_ + c] = (__bf16)0.f;
        }
    }
    __syncthreads();

    // message MLP
    gemm128(inA, KINP_, sW1, sb1, KINP_, true,  hB, nullptr, nullptr, nullptr, 0, tid);
    __syncthreads();
    gemm128(hB, 64, sW2, sb2, 64, true,  hC, nullptr, nullptr, nullptr, 0, tid);
    __syncthreads();
    gemm128(hC, 64, sW3, sb3, 64, false, hD, agg_m, sdst, nullptr, 0, tid);  // m + scatter
    __syncthreads();
    // upd_pos MLP on m
    gemm128(hD, 64, sWp1, sbp1, 64, true, hB, nullptr, nullptr, nullptr, 0, tid);
    __syncthreads();
    gemm128(hB, 64, sWp2, sbp2, 64, true, hC, nullptr, nullptr, nullptr, 0, tid);
    __syncthreads();
    // scalar head + pos message scatter
    if (tid < 128) {
        float acc = bp3[0];
        for (int k = 0; k < 64; ++k)
            acc += (float)hC[tid * 64 + k] * (float)swp3[k];
        int d = sdst[tid];
#pragma unroll
        for (int dd = 0; dd < 3; ++dd)
            atomicAdd(&agg_pos[(size_t)d * 3 + dd], acc * sdiff[tid * 3 + dd]);
    }
}

// ---------------------------------------------------------------------------
// Node kernel: 128 nodes per block. upd_vel (2 GEMMs + dot), vel/pos update,
// upd_feat (3 GEMMs) -> x_new.
// ---------------------------------------------------------------------------
__global__ __launch_bounds__(256) void node_kernel(
    const __bf16* __restrict__ xs_bf,
    const float* __restrict__ pos2, const float* __restrict__ vel2,
    const float* __restrict__ agg_m, const float* __restrict__ agg_pos,
    const __bf16* __restrict__ wv1, const __bf16* __restrict__ wv2,
    const __bf16* __restrict__ wv3g,
    const float* __restrict__ bv1, const float* __restrict__ bv2,
    const float* __restrict__ bv3,
    const __bf16* __restrict__ wf1, const __bf16* __restrict__ wf2,
    const __bf16* __restrict__ wf3,
    const float* __restrict__ bf1, const float* __restrict__ bf2,
    const float* __restrict__ bf3,
    float* __restrict__ xout, float* __restrict__ posout, float* __restrict__ velout)
{
    extern __shared__ char smem[];
    char* p = smem;
    float* sbv1 = (float*)p; p += 256;
    float* sbv2 = (float*)p; p += 256;
    float* sbf1 = (float*)p; p += 256;
    float* sbf2 = (float*)p; p += 256;
    float* sbf3 = (float*)p; p += 256;
    __bf16* inA  = (__bf16*)p; p += 128 * 128 * 2;
    __bf16* sWv1 = (__bf16*)p; p += 4096 * 2;    // transposed [64][64]
    __bf16* sWv2 = (__bf16*)p; p += 4096 * 2;
    __bf16* swv3 = (__bf16*)p; p += 64 * 2;
    __bf16* sWf1 = (__bf16*)p; p += 8192 * 2;    // transposed [64][128]
    __bf16* sWf2 = (__bf16*)p; p += 4096 * 2;
    __bf16* sWf3 = (__bf16*)p; p += 4096 * 2;
    __bf16* hB   = (__bf16*)p; p += 128 * 64 * 2;
    __bf16* hC   = (__bf16*)p; p += 128 * 64 * 2;

    int tid = threadIdx.x;
    int n0  = blockIdx.x * 128;

    for (int i = tid; i < 8192 / 8; i += 256)
        ((v8bf*)sWf1)[i] = ((const v8bf*)wf1)[i];
    for (int i = tid; i < 4096 / 8; i += 256) {
        ((v8bf*)sWv1)[i] = ((const v8bf*)wv1)[i];
        ((v8bf*)sWv2)[i] = ((const v8bf*)wv2)[i];
        ((v8bf*)sWf2)[i] = ((const v8bf*)wf2)[i];
        ((v8bf*)sWf3)[i] = ((const v8bf*)wf3)[i];
    }
    if (tid < 64) {
        swv3[tid] = wv3g[tid];
        sbv1[tid] = bv1[tid]; sbv2[tid] = bv2[tid];
        sbf1[tid] = bf1[tid]; sbf2[tid] = bf2[tid]; sbf3[tid] = bf3[tid];
    }

    // build [x | agg_m] 128x128 bf16 tile
    {
        int r  = tid >> 1;
        int hb = tid & 1;
        size_t node = (size_t)(n0 + r);
        if (hb == 0) {
            const v8bf* srow = (const v8bf*)(xs_bf + node * 64);
            v8bf* drow = (v8bf*)(inA + r * 128);
#pragma unroll
            for (int j = 0; j < 8; ++j) drow[j] = srow[j];
        } else {
            for (int j = 0; j < 64; ++j)
                inA[r * 128 + 64 + j] = (__bf16)agg_m[node * 64 + j];
        }
    }
    __syncthreads();

    // upd_vel MLP (uses only first 64 cols of inA)
    gemm128(inA, 128, sWv1, sbv1, 64, true, hB, nullptr, nullptr, nullptr, 0, tid);
    __syncthreads();
    gemm128(hB, 64, sWv2, sbv2, 64, true, hC, nullptr, nullptr, nullptr, 0, tid);
    __syncthreads();
    if (tid < 128) {
        float acc = bv3[0];
        for (int k = 0; k < 64; ++k)
            acc += (float)hC[tid * 64 + k] * (float)swv3[k];
        size_t node = (size_t)(n0 + tid);
#pragma unroll
        for (int d = 0; d < 3; ++d) {
            float vn = acc * vel2[node * 3 + d] + agg_pos[node * 3 + d];
            velout[node * 3 + d] = vn;
            posout[node * 3 + d] = pos2[node * 3 + d] + vn;
        }
    }
    __syncthreads();

    // upd_feat MLP -> x_new (direct f32 store to d_out)
    gemm128(inA, 128, sWf1, sbf1, 128, true, hB, nullptr, nullptr, nullptr, 0, tid);
    __syncthreads();
    gemm128(hB, 64, sWf2, sbf2, 64, true, hC, nullptr, nullptr, nullptr, 0, tid);
    __syncthreads();
    gemm128(hC, 64, sWf3, sbf3, 64, false, nullptr, nullptr, nullptr, xout, n0, tid);
}

// ---------------------------------------------------------------------------
// Host-side launch
// ---------------------------------------------------------------------------
extern "C" void kernel_launch(void* const* d_in, const int* in_sizes, int n_in,
                              void* d_out, int out_size, void* d_ws, size_t ws_size,
                              hipStream_t stream) {
    (void)in_sizes; (void)n_in; (void)out_size; (void)ws_size;

    // d_in order (setup_inputs dict order, params flattened):
    // 0:x 1:pos 2:vel 3:edge_index 4:edge_attr 5:wsr 6:wsi 7:wvr 8:wvi
    // 9..14:  message  W1,b1,W2,b2,W3,b3
    // 15..20: upd_feat W1,b1,W2,b2,W3,b3
    // 21..26: upd_pos  W1,b1,W2,b2,W3,b3
    // 27..32: upd_vel  W1,b1,W2,b2,W3,b3
    const float* x    = (const float*)d_in[0];
    const float* pos  = (const float*)d_in[1];
    const float* vel  = (const float*)d_in[2];
    const int*   eidx = (const int*)  d_in[3];
    const float* eatt = (const float*)d_in[4];
    const float* wsr  = (const float*)d_in[5];
    const float* wsi  = (const float*)d_in[6];
    const float* wvr  = (const float*)d_in[7];
    const float* wvi  = (const float*)d_in[8];

    // workspace carve (f32 section first, then bf16)
    float* xs_f    = (float*)d_ws;
    float* pos2    = xs_f    + (size_t)NODES_ * F_;       // 240000
    float* vel2    = pos2    + (size_t)NODES_ * 3;        // 240000
    float* center  = vel2    + (size_t)NODES_ * 3;        // 24 (pad to 32)
    float* agg_m   = center  + 32;                        // 5,120,000
    float* agg_pos = agg_m   + (size_t)NODES_ * H_;       // 240000
    float* fend    = agg_pos + (size_t)NODES_ * 3;
    __bf16* xs_bf  = (__bf16*)fend;
    __bf16* wm1 = xs_bf + (size_t)NODES_ * F_;   // [64][160] transposed
    __bf16* wm2 = wm1 + KINP_ * 64;              // [64][64] transposed
    __bf16* wm3 = wm2 + 4096;
    __bf16* wp1 = wm3 + 4096;
    __bf16* wp2 = wp1 + 4096;
    __bf16* wp3 = wp2 + 4096;                    // [64] vector
    __bf16* wv1 = wp3 + 64;
    __bf16* wv2 = wv1 + 4096;
    __bf16* wv3 = wv2 + 4096;                    // [64] vector
    __bf16* wf1 = wv3 + 64;                      // [64][128] transposed
    __bf16* wf2 = wf1 + 8192;
    __bf16* wf3 = wf2 + 4096;

    float* xout   = (float*)d_out;
    float* posout = xout   + (size_t)NODES_ * F_;
    float* velout = posout + (size_t)NODES_ * 3;

    // zero the aggregation buffers (agg_m and agg_pos are contiguous)
    {
        int zc = NODES_ * H_ + NODES_ * 3;
        zero_kernel<<<(zc + 255) / 256, 256, 0, stream>>>(agg_m, zc);
    }

    // bf16 transposed weight conversion
    cvt_t_kernel<<<40, 256, 0, stream>>>((const float*)d_in[9],  wm1, KIN_, KINP_);
    cvt_t_kernel<<<16, 256, 0, stream>>>((const float*)d_in[11], wm2, 64, 64);
    cvt_t_kernel<<<16, 256, 0, stream>>>((const float*)d_in[13], wm3, 64, 64);
    cvt_t_kernel<<<32, 256, 0, stream>>>((const float*)d_in[15], wf1, 128, 128);
    cvt_t_kernel<<<16, 256, 0, stream>>>((const float*)d_in[17], wf2, 64, 64);
    cvt_t_kernel<<<16, 256, 0, stream>>>((const float*)d_in[19], wf3, 64, 64);
    cvt_t_kernel<<<16, 256, 0, stream>>>((const float*)d_in[21], wp1, 64, 64);
    cvt_t_kernel<<<16, 256, 0, stream>>>((const float*)d_in[23], wp2, 64, 64);
    cvt_kernel  <<<1,  256, 0, stream>>>((const float*)d_in[25], wp3, 64);
    cvt_t_kernel<<<16, 256, 0, stream>>>((const float*)d_in[27], wv1, 64, 64);
    cvt_t_kernel<<<16, 256, 0, stream>>>((const float*)d_in[29], wv2, 64, 64);
    cvt_kernel  <<<1,  256, 0, stream>>>((const float*)d_in[31], wv3, 64);

    // spectral stage
    center_kernel<<<T_ * 3, 256, 0, stream>>>(pos, center);
    spectral_x_kernel<<<N_, 256, 0, stream>>>(x, wsr, wsi, xs_f, xs_bf);
    spectral_vec_kernel<<<(N_ * 3 + 255) / 256, 256, 0, stream>>>(
        pos, vel, center, wvr, wvi, pos2, vel2);

    // edge stage (WMMA): 5000 blocks x 128 edges
    constexpr size_t EDGE_SMEM =
        5 * 256 + 128 * 3 * 4 + 128 * 4 + 128 * 4 + 128 * 4 +          // f32/int
        (128 * KINP_ + KINP_ * 64 + 4 * 4096 + 64 + 3 * 128 * 64) * 2; // bf16
    edge_kernel<<<E_ / 128, 256, EDGE_SMEM, stream>>>(
        eidx, eatt, xs_bf, pos2,
        wm1, wm2, wm3, wp1, wp2, wp3,
        (const float*)d_in[10], (const float*)d_in[12], (const float*)d_in[14],
        (const float*)d_in[22], (const float*)d_in[24], (const float*)d_in[26],
        agg_m, agg_pos);

    // node stage (WMMA): 625 blocks x 128 nodes
    constexpr size_t NODE_SMEM =
        5 * 256 +
        (128 * 128 + 2 * 4096 + 64 + 8192 + 2 * 4096 + 2 * 128 * 64) * 2;
    node_kernel<<<NODES_ / 128, 256, NODE_SMEM, stream>>>(
        xs_bf, pos2, vel2, agg_m, agg_pos,
        wv1, wv2, wv3,
        (const float*)d_in[28], (const float*)d_in[30], (const float*)d_in[32],
        wf1, wf2, wf3,
        (const float*)d_in[16], (const float*)d_in[18], (const float*)d_in[20],
        xout, posout, velout);
}
